// MultiHeadAttention_78331613544953
// MI455X (gfx1250) — matrix operand order repaired and checked by reading, not compiled, not run on hardware
//
#include <hip/hip_runtime.h>

// MI455X / gfx1250 multi-head attention forward, f16 end-to-end internally.
// - All matrix math: v_wmma_f32_16x16x32_f16 (wave32 WMMA, f32 accum).
// - GEMM + attention K tiles staged with GLOBAL_LOAD_ASYNC_TO_LDS_B128
//   (ASYNCcnt), ping-pong LDS, one barrier per step.
// - All WMMA fragments are 16-byte ds_load_b128 / global_load_b128 pairs per
//   the k = 16*half + s order used here (A and B alike: contiguous 16-half
//   K-run/lane when the tile is stored K-contiguous).
// - Attention: 4 waves/block share one (b,h): K/V chunks staged cooperatively
//   (4x less L2 traffic), V pre-transposed in LDS so PV fragments vectorize.

typedef __attribute__((ext_vector_type(16))) _Float16 v16h;
typedef __attribute__((ext_vector_type(8)))  _Float16 v8h;
typedef __attribute__((ext_vector_type(4)))  _Float16 v4h;
typedef __attribute__((ext_vector_type(8)))  float    v8f;
typedef __attribute__((ext_vector_type(4)))  float    v4f;

#define NEGBIG (-1e9f)

union U16h { v16h v; v8h h[2]; };

__device__ __forceinline__ v8f wmma_f16(v16h a, v16h b, v8f c) {
  return __builtin_amdgcn_wmma_f32_16x16x32_f16(false, a, false, b,
                                                (short)0, c, false, false);
}

__device__ __forceinline__ v8f vzero8() {
  return (v8f){0.f,0.f,0.f,0.f,0.f,0.f,0.f,0.f};
}

__device__ __forceinline__ v4h cvt4(v4f f) {
  v4h r; r[0] = (_Float16)f[0]; r[1] = (_Float16)f[1];
         r[2] = (_Float16)f[2]; r[3] = (_Float16)f[3];
  return r;
}

// Async 16-byte global -> LDS copy (CDNA5 GLOBAL_LOAD_ASYNC_TO_LDS_B128).
__device__ __forceinline__ void async_cp16(unsigned lds, const _Float16* g) {
  asm volatile("global_load_async_to_lds_b128 %0, %1, off"
               :: "v"(lds), "v"((unsigned long long)(uintptr_t)g) : "memory");
}
__device__ __forceinline__ void wait_async0() {
  asm volatile("s_wait_asynccnt 0x0" ::: "memory");
}

// ---------------------------------------------------------------------------
// Elementwise fp32 -> f16 (x conversion). 4 elements/thread.
// ---------------------------------------------------------------------------
__global__ __launch_bounds__(256)
void cvt_f32_to_f16(const float* __restrict__ in, _Float16* __restrict__ out)
{
  size_t i = ((size_t)blockIdx.x * 256 + threadIdx.x) * 4;
  *(v4h*)&out[i] = cvt4(*(const v4f*)&in[i]);
}

// ---------------------------------------------------------------------------
// W[K,N] fp32 -> Wt[N,K] f16 (transpose+convert), 32x32 LDS tiles.
// ---------------------------------------------------------------------------
__global__ __launch_bounds__(256)
void transpose_cvt(const float* __restrict__ W, _Float16* __restrict__ Wt,
                   int K, int N)
{
  __shared__ _Float16 T[32][33];
  const int t  = threadIdx.x;
  const int n0 = blockIdx.x * 32;
  const int k0 = blockIdx.y * 32;

  {
    int row = t >> 3, c4 = (t & 7) * 4;
    v4f w = *(const v4f*)&W[(size_t)(k0 + row) * N + n0 + c4];
#pragma unroll
    for (int q = 0; q < 4; ++q) T[row][c4 + q] = (_Float16)w[q];
  }
  __syncthreads();
  {
    int nr = t >> 3, k4 = (t & 7) * 4;
    v4h o;
#pragma unroll
    for (int q = 0; q < 4; ++q) o[q] = T[k4 + q][nr];
    *(v4h*)&Wt[(size_t)(n0 + nr) * K + k0 + k4] = o;
  }
}

// ---------------------------------------------------------------------------
// C[M,N] = A[M,K] * Bt[N,K]^T + bias[N]; f16 inputs (Bt pre-transposed),
// f32 accum, output f16 or f32. 256 threads (8 waves), tile 128x128,
// K-step 32. Both tiles staged via async-to-LDS, ping-pong, 1 barrier/step.
// ---------------------------------------------------------------------------
template <bool HALF_OUT>
__global__ __launch_bounds__(256)
void gemm_f16(const _Float16* __restrict__ A, const _Float16* __restrict__ Bt,
              const float* __restrict__ bias, void* __restrict__ Cv,
              int M, int N, int K)
{
  __shared__ __align__(16) _Float16 As[2][128 * 40];   // [buf][m][k]
  __shared__ __align__(16) _Float16 Bs[2][128 * 40];   // [buf][n][k]

  const int tid  = threadIdx.x;
  const int wave = tid >> 5;
  const int lane = tid & 31;
  const int half = (lane >> 4) & 1;
  const int lm   = lane & 15;

  const int m_blk = blockIdx.y * 128;
  const int n_blk = blockIdx.x * 128;
  const int mloc  = wave << 4;

  const unsigned asB = (unsigned)(uintptr_t)&As[0][0];
  const unsigned bsB = (unsigned)(uintptr_t)&Bs[0][0];

  auto stage = [&](int bb, int kk) {
#pragma unroll
    for (int i = 0; i < 2; ++i) {               // 512 16B chunks of A
      int idx = i * 256 + tid;
      int r = idx >> 2, c8 = (idx & 3) * 8;
      async_cp16(asB + bb * 10240 + (r * 40 + c8) * 2,
                 A + (size_t)(m_blk + r) * K + kk + c8);
    }
#pragma unroll
    for (int i = 0; i < 2; ++i) {               // 512 16B chunks of B
      int idx = i * 256 + tid;
      int n = idx >> 2, c8 = (idx & 3) * 8;
      async_cp16(bsB + bb * 10240 + (n * 40 + c8) * 2,
                 Bt + (size_t)(n_blk + n) * K + kk + c8);
    }
  };

  v8f acc[8];
#pragma unroll
  for (int j = 0; j < 8; ++j) acc[j] = vzero8();

  stage(0, 0);
  wait_async0();
  __syncthreads();

  int buf = 0;
  for (int kt = 0; kt < K; kt += 32) {
    const bool more = (kt + 32) < K;
    if (more) stage(buf ^ 1, kt + 32);          // in flight during compute

    U16h aF;
    {
      const _Float16* ap = &As[buf][(mloc + lm) * 40];
      aF.h[0] = *(const v8h*)(ap + half * 16);
      aF.h[1] = *(const v8h*)(ap + half * 16 + 8);
    }
    U16h bF[2];
    const _Float16* bbase = &Bs[buf][lm * 40 + half * 16];
    bF[0].h[0] = *(const v8h*)(bbase);
    bF[0].h[1] = *(const v8h*)(bbase + 8);
#pragma unroll
    for (int j = 0; j < 8; ++j) {
      if (j < 7) {
        const _Float16* bp = bbase + (j + 1) * 16 * 40;
        bF[(j + 1) & 1].h[0] = *(const v8h*)(bp);
        bF[(j + 1) & 1].h[1] = *(const v8h*)(bp + 8);
      }
      acc[j] = wmma_f16(aF.v, bF[j & 1].v, acc[j]);
    }

    if (more) {
      wait_async0();
      __syncthreads();
      buf ^= 1;
    }
  }

#pragma unroll
  for (int j = 0; j < 8; ++j) {
    int n = n_blk + j * 16 + lm;
    float bv = bias[n];
#pragma unroll
    for (int r = 0; r < 8; ++r) {
      int m = m_blk + mloc + r + 8 * half;
      float val = acc[j][r] + bv;
      if constexpr (HALF_OUT)
        ((_Float16*)Cv)[(size_t)m * N + n] = (_Float16)val;
      else
        ((float*)Cv)[(size_t)m * N + n] = val;
    }
  }
}

// ---------------------------------------------------------------------------
// Flash-attention: block = 4 waves = 4 consecutive 16-row q tiles of ONE
// (b,h). K/V 32-key chunks staged cooperatively into ping-pong LDS
// (K via async-to-LDS; V transposed during staging). Causal mask analytic.
// qkv: [B,S,3D] f16; att: [B,H,S,64] f16 (== reference reshape layout).
// ---------------------------------------------------------------------------
__global__ __launch_bounds__(128)
void attn_fwd(const _Float16* __restrict__ qkv, _Float16* __restrict__ att,
              int S)
{
  __shared__ __align__(16) _Float16 Kl[2][32 * 72];  // [buf][key][d] d-contig
  __shared__ __align__(16) _Float16 Vt[2][64 * 40];  // [buf][d][kk] kk-contig
  __shared__ __align__(16) _Float16 Pb[4][16 * 32];  // per-wave P (16q x 32k)

  const int tid  = threadIdx.x;
  const int wave = tid >> 5;
  const int lane = tid & 31;
  const int half = (lane >> 4) & 1;
  const int lm   = lane & 15;

  const int gw = blockIdx.x * 4 + wave;
  const int qt = gw & 127;                 // S/16 = 128 q-tiles per (b,h)
  const int bh = gw >> 7;
  const int q0 = qt << 4;
  const int b  = bh >> 4, h = bh & 15;     // H = 16
  const int qlast = q0 + 15;               // per-wave causal bound
  const int kend  = (blockIdx.x & 31) * 64 + 63;   // block causal bound

  const _Float16* base = qkv + (size_t)b * S * 3072 + h * 192;
  const unsigned klB = (unsigned)(uintptr_t)&Kl[0][0];

  // Stage K chunk (32 keys x 64 d) via async copies: 2 x 16B per thread.
  auto stageK = [&](int bb, int k0) {
#pragma unroll
    for (int i = 0; i < 2; ++i) {
      int id = i * 128 + tid;                     // 256 chunks
      int key = id >> 3, c8 = (id & 7) * 8;
      async_cp16(klB + bb * (32 * 72 * 2) + (key * 72 + c8) * 2,
                 base + (size_t)(k0 + key) * 3072 + 64 + c8);
    }
  };
  // Stage V chunk transposed: 2 coalesced b128 reads + 16 b16 LDS writes.
  auto stageV = [&](int bb, int k0) {
    int key = tid >> 2, c16 = (tid & 3) * 16;
    const _Float16* vp = base + (size_t)(k0 + key) * 3072 + 128 + c16;
    v8h v0 = *(const v8h*)(vp);
    v8h v1 = *(const v8h*)(vp + 8);
#pragma unroll
    for (int q = 0; q < 8; ++q) {
      Vt[bb][(c16 + q) * 40 + key]     = v0[q];
      Vt[bb][(c16 + 8 + q) * 40 + key] = v1[q];
    }
  };

  // Q fragments: f16 -> each 8-half K-run is one global b128 load.
  U16h aQ0, aQ1;
  {
    const _Float16* qp = base + (size_t)(q0 + lm) * 3072;
    aQ0.h[0] = *(const v8h*)(qp + half * 16);
    aQ0.h[1] = *(const v8h*)(qp + half * 16 + 8);
    aQ1.h[0] = *(const v8h*)(qp + 32 + half * 16);
    aQ1.h[1] = *(const v8h*)(qp + 40 + half * 16);
  }

  float m_i[8], l_i[8];
#pragma unroll
  for (int r = 0; r < 8; ++r) { m_i[r] = -1e30f; l_i[r] = 0.f; }
  v8f o[4];
#pragma unroll
  for (int j = 0; j < 4; ++j) o[j] = vzero8();

  stageK(0, 0); stageV(0, 0);
  wait_async0();
  __syncthreads();

  int buf = 0;
  for (int k0 = 0; k0 <= kend; k0 += 32) {   // block-uniform trip count
    const bool more = (k0 + 32) <= kend;
    if (more) { stageK(buf ^ 1, k0 + 32); stageV(buf ^ 1, k0 + 32); }

    if (k0 <= qlast) {                       // wave-uniform causal guard
      v8f sc[2];
#pragma unroll
      for (int t = 0; t < 2; ++t) {
        const int kb = k0 + t * 16;
        if (kb <= qlast) {
          U16h bK0, bK1;                     // d-runs from LDS K tile
          const _Float16* kp = &Kl[buf][(t * 16 + lm) * 72 + half * 16];
          bK0.h[0] = *(const v8h*)(kp);
          bK0.h[1] = *(const v8h*)(kp + 8);
          bK1.h[0] = *(const v8h*)(kp + 32);
          bK1.h[1] = *(const v8h*)(kp + 40);
          v8f c = vzero8();
          c = wmma_f16(aQ0.v, bK0.v, c);
          c = wmma_f16(aQ1.v, bK1.v, c);
          const int key = kb + lm;
#pragma unroll
          for (int r = 0; r < 8; ++r) {
            int qrow = q0 + r + 8 * half;
            sc[t][r] = c[r] * 0.125f + ((key > qrow) ? NEGBIG : 0.f);
          }
        } else {
#pragma unroll
          for (int r = 0; r < 8; ++r) sc[t][r] = NEGBIG;
        }
      }

      // Online softmax (per-row stats; rows live across 16-lane halves).
      float alpha[8];
#pragma unroll
      for (int r = 0; r < 8; ++r) {
        float v = fmaxf(sc[0][r], sc[1][r]);
#pragma unroll
        for (int off = 8; off >= 1; off >>= 1)
          v = fmaxf(v, __shfl_xor(v, off, 16));
        float mn = fmaxf(m_i[r], v);
        alpha[r] = __expf(m_i[r] - mn);
        float p0 = __expf(sc[0][r] - mn);
        float p1 = __expf(sc[1][r] - mn);
        sc[0][r] = p0; sc[1][r] = p1;
        float rs = p0 + p1;
#pragma unroll
        for (int off = 8; off >= 1; off >>= 1)
          rs += __shfl_xor(rs, off, 16);
        l_i[r] = l_i[r] * alpha[r] + rs;
        m_i[r] = mn;
      }

#pragma unroll
      for (int j = 0; j < 4; ++j)
#pragma unroll
        for (int r = 0; r < 8; ++r)
          o[j][r] *= alpha[r];

      // C-layout -> A-layout transpose of P through wave-private LDS slice.
#pragma unroll
      for (int t = 0; t < 2; ++t)
#pragma unroll
        for (int r = 0; r < 8; ++r)
          Pb[wave][(r + 8 * half) * 32 + t * 16 + lm] = (_Float16)sc[t][r];
      asm volatile("s_wait_dscnt 0x0" ::: "memory");

      U16h aP;
      {
        const _Float16* pp = &Pb[wave][lm * 32];
        aP.h[0] = *(const v8h*)(pp + half * 16);
        aP.h[1] = *(const v8h*)(pp + half * 16 + 8);
      }

      // O += P(16x32) x V(32x64): V fragments now 2x ds_load_b128 from Vt.
#pragma unroll
      for (int j = 0; j < 4; ++j) {
        U16h bV;
        const _Float16* vp = &Vt[buf][(j * 16 + lm) * 40 + half * 16];
        bV.h[0] = *(const v8h*)(vp);
        bV.h[1] = *(const v8h*)(vp + 8);
        o[j] = wmma_f16(aP.v, bV.v, o[j]);
      }
    }

    if (more) {
      wait_async0();
      __syncthreads();
      buf ^= 1;
    }
  }

#pragma unroll
  for (int r = 0; r < 8; ++r) {
    float inv = 1.f / l_i[r];
    int row = q0 + r + 8 * half;
#pragma unroll
    for (int j = 0; j < 4; ++j)
      att[((size_t)bh * S + row) * 64 + j * 16 + lm] =
          (_Float16)(o[j][r] * inv);
  }
}

// ---------------------------------------------------------------------------
extern "C" void kernel_launch(void* const* d_in, const int* in_sizes, int n_in,
                              void* d_out, int out_size, void* d_ws, size_t ws_size,
                              hipStream_t stream)
{
  const float* x     = (const float*)d_in[0];  // [2,2048,1024]
  const float* W_qkv = (const float*)d_in[1];  // [1024,3072]
  const float* b_qkv = (const float*)d_in[2];  // [3072]
  const float* W_out = (const float*)d_in[3];  // [1024,1024]
  const float* b_out = (const float*)d_in[4];  // [1024]
  // d_in[5] = causal mask, reproduced analytically in attn_fwd

  const int BS = 2 * 2048;   // 4096 rows (B*S)
  const int Dm = 1024;
  const int N1 = 3 * Dm;     // 3072

  // f16 workspace (~50 MB total, L2-resident).
  _Float16* qkvh = (_Float16*)d_ws;                    // [4096,3072]
  _Float16* atth = qkvh + (size_t)BS * N1;             // [B,H,S,64]
  _Float16* xh   = atth + (size_t)BS * Dm;             // [4096,1024]
  _Float16* wqT  = xh   + (size_t)BS * Dm;             // [3072,1024] = Wqkv^T
  _Float16* woT  = wqT  + (size_t)N1 * Dm;             // [1024,1024] = Wout^T

  // 0) One-time conversions: x -> f16; weights -> f16 transposed (K-contig).
  cvt_f32_to_f16<<<dim3((BS * Dm) / 1024), dim3(256), 0, stream>>>(x, xh);
  transpose_cvt<<<dim3(N1 / 32, Dm / 32), dim3(256), 0, stream>>>(
      W_qkv, wqT, Dm, N1);
  transpose_cvt<<<dim3(Dm / 32, Dm / 32), dim3(256), 0, stream>>>(
      W_out, woT, Dm, Dm);

  // 1) Fused QKV projection (f16 out)
  gemm_f16<true><<<dim3(N1 / 128, BS / 128), dim3(256), 0, stream>>>(
      xh, wqT, b_qkv, qkvh, BS, N1, Dm);

  // 2) Causal flash attention: B*H*(S/16) = 4096 waves, 4 per block
  attn_fwd<<<dim3(4096 / 4), dim3(128), 0, stream>>>(qkvh, atth, 2048);

  // 3) Output projection (raw-reshape layout is already [4096,1024]; fp32 out)
  gemm_f16<false><<<dim3(Dm / 128, BS / 128), dim3(256), 0, stream>>>(
      atth, woT, b_out, d_out, BS, Dm, Dm);
}
